// QKNormMHA_26731876450483
// MI455X (gfx1250) — matrix operand order repaired and checked by reading, not compiled, not run on hardware
//
#include <hip/hip_runtime.h>

// ---------------------------------------------------------------------------
// CDNA5 (gfx1250) wave32 WMMA bf16 path.
// D = A(16x32 bf16) x B(32x16 bf16) + C(16x16 f32)
// ---------------------------------------------------------------------------

typedef __attribute__((ext_vector_type(16))) __bf16 bf16x16;
typedef __attribute__((ext_vector_type(8)))  float  f32x8;
typedef __attribute__((ext_vector_type(4)))  unsigned int u32x4;
typedef __attribute__((ext_vector_type(8)))  int i32x8;
typedef __attribute__((ext_vector_type(4)))  int i32x4;

union Frag {
    bf16x16 v;
    uint4   q[2];
};

static __device__ __forceinline__ f32x8 wmma_bf16(const Frag& a, const Frag& b, f32x8 c) {
    return __builtin_amdgcn_wmma_f32_16x16x32_bf16(false, a.v, false, b.v,
                                                   (short)0, c, false, false);
}

// ---------------------------------------------------------------------------
// Tensor Data Mover: DMA one [rows x 128] bf16 tile (row stride 128 elements)
// from global memory into LDS. D# bitfields per CDNA5 ISA.
// Issued by one wave; completion via TENSORcnt.
// ---------------------------------------------------------------------------
static __device__ __forceinline__ void tdm_load_tile_b16_128w(
        const __bf16* gsrc, unsigned lds_byte_off, int rows)
{
    unsigned long long ga = (unsigned long long)(const void*)gsrc;
    u32x4 g0;
    g0[0] = 1u;                                   // count=1, user, no gather
    g0[1] = lds_byte_off;                         // lds_addr
    g0[2] = (unsigned)(ga & 0xFFFFFFFFu);         // global_addr[31:0]
    g0[3] = (unsigned)((ga >> 32) & 0x01FFFFFFu)  // global_addr[56:32]
          | (2u << 30);                           // type = 2 ("image")
    i32x8 g1;
    g1[0] = 0x00010000;          // wg_mask=0, data_size=1 (2 bytes)
    g1[1] = (int)(128u << 16);   // ab_addr=0 | tensor_dim0[15:0]=128
    g1[2] = (int)(2048u << 16);  // tensor_dim0[31:16]=0 | tensor_dim1[15:0]=2048
    g1[3] = (int)(128u << 16);   // tensor_dim1[31:16]=0 | tile_dim0=128
    g1[4] = rows;                // tile_dim1=rows | tile_dim2=0
    g1[5] = 128;                 // tensor_dim0_stride[31:0] = 128 elements
    g1[6] = 0;                   // stride0 hi | tensor_dim1_stride lo (unused, 2D)
    g1[7] = 0;
    i32x4 z4 = {0, 0, 0, 0};
#if __clang_major__ >= 23
    i32x8 z8 = {0, 0, 0, 0, 0, 0, 0, 0};
    __builtin_amdgcn_tensor_load_to_lds(g0, g1, z4, z4, z8, 0);
#else
    __builtin_amdgcn_tensor_load_to_lds(g0, g1, z4, z4, 0);
#endif
}

// ---------------------------------------------------------------------------
// fp32 -> bf16 conversion (grid-stride)
// ---------------------------------------------------------------------------
__global__ void __launch_bounds__(256)
conv_f32_bf16(const float* __restrict__ in, __bf16* __restrict__ out, int n) {
    for (int i = blockIdx.x * 256 + threadIdx.x; i < n; i += gridDim.x * 256)
        out[i] = (__bf16)in[i];
}

// ---------------------------------------------------------------------------
// Tiled bf16 GEMM: C[m,n] = sum_k A[m,k] * Bm[n,k]   (Bm is torch [out,in])
// Block: 256 threads (8 waves), tile 128(M) x 128(N), K-step 32,
// double-buffered LDS staging + L2 prefetch 2 steps ahead.
// Wave grid 4(M) x 2(N); each wave: 2x4 subtiles of 16x16.
//
// FUSED=true (QKV projection): N-tile == one head slice of q/k/v.
//   Epilogue does RMSNorm over the 128-wide tile (LDS ds_add_f32 reduction)
//   and writes bf16 q/k/v in [B,H,S,HD] layout.
// FUSED=false: plain fp32 store (output projection).
// ---------------------------------------------------------------------------
template <bool FUSED>
__global__ void __launch_bounds__(256)
gemm_bf16_wmma(const __bf16* __restrict__ A, const __bf16* __restrict__ Bm,
               float* __restrict__ Cout,
               __bf16* __restrict__ qOut, __bf16* __restrict__ kOut,
               __bf16* __restrict__ vOut,
               const float* __restrict__ qg, const float* __restrict__ kg,
               int M, int N, int K)
{
    __shared__ __bf16 As[2][128 * 32];
    __shared__ __bf16 Bs[2][128 * 32];
    __shared__ float  ssq[128];
    __shared__ float  gs[128];

    const int tid  = threadIdx.x;
    const int lane = tid & 31;
    const int wid  = tid >> 5;
    const int wm   = wid & 3;   // wave M index (0..3)
    const int wn   = wid >> 2;  // wave N index (0..1)
    const int ln   = lane & 15;
    const int hi   = lane >> 4;
    const int gm0  = blockIdx.x * 128;
    const int gn0  = blockIdx.y * 128;

    const f32x8 zero = {0.f, 0.f, 0.f, 0.f, 0.f, 0.f, 0.f, 0.f};
    f32x8 acc[2][4];
#pragma unroll
    for (int mi = 0; mi < 2; ++mi)
#pragma unroll
        for (int ni = 0; ni < 4; ++ni) acc[mi][ni] = zero;

    // global->LDS staging: 2 threads per row, 32B each (2 x b128)
    const int rowLd  = tid >> 1;
    const int halfLd = tid & 1;
    const __bf16* aSrc = A  + (size_t)(gm0 + rowLd) * K + halfLd * 16;
    const __bf16* bSrc = Bm + (size_t)(gn0 + rowLd) * K + halfLd * 16;
    const int dOff = rowLd * 32 + halfLd * 16;

    const int nsteps = K >> 5;

    // prologue: stage tile 0 into buffer 0
    ((uint4*)(As[0] + dOff))[0] = ((const uint4*)(aSrc))[0];
    ((uint4*)(As[0] + dOff))[1] = ((const uint4*)(aSrc))[1];
    ((uint4*)(Bs[0] + dOff))[0] = ((const uint4*)(bSrc))[0];
    ((uint4*)(Bs[0] + dOff))[1] = ((const uint4*)(bSrc))[1];

    for (int s = 0; s < nsteps; ++s) {
        __syncthreads();   // buffer s&1 visible; buffer (s+1)&1 free
        const int cur = s & 1;
        if (s + 1 < nsteps) {
            const int nxt = (s + 1) & 1;
            const int k1 = (s + 1) << 5;
            ((uint4*)(As[nxt] + dOff))[0] = ((const uint4*)(aSrc + k1))[0];
            ((uint4*)(As[nxt] + dOff))[1] = ((const uint4*)(aSrc + k1))[1];
            ((uint4*)(Bs[nxt] + dOff))[0] = ((const uint4*)(bSrc + k1))[0];
            ((uint4*)(Bs[nxt] + dOff))[1] = ((const uint4*)(bSrc + k1))[1];
        }
        if (s + 2 < nsteps) {      // L2 prefetch (global_prefetch_b8)
            __builtin_prefetch(aSrc + ((s + 2) << 5), 0, 1);
            __builtin_prefetch(bSrc + ((s + 2) << 5), 0, 1);
        }

        Frag a[2];
#pragma unroll
        for (int mi = 0; mi < 2; ++mi) {
            const __bf16* p = As[cur] + (wm * 32 + mi * 16 + ln) * 32 + hi * 8;
            a[mi].q[0] = *(const uint4*)p;          // K = hi*8   .. hi*8+7
            a[mi].q[1] = *(const uint4*)(p + 16);   // K = hi*8+16.. hi*8+23
        }
#pragma unroll
        for (int ni = 0; ni < 4; ++ni) {
            Frag b;
            const __bf16* p = Bs[cur] + (wn * 64 + ni * 16 + ln) * 32 + hi * 8;
            b.q[0] = *(const uint4*)p;              // K = hi*8   .. hi*8+7
            b.q[1] = *(const uint4*)(p + 16);       // K = hi*8+16.. hi*8+23
#pragma unroll
            for (int mi = 0; mi < 2; ++mi)
                acc[mi][ni] = wmma_bf16(a[mi], b, acc[mi][ni]);
        }
    }
    __syncthreads();

    if constexpr (!FUSED) {
#pragma unroll
        for (int mi = 0; mi < 2; ++mi)
#pragma unroll
            for (int ni = 0; ni < 4; ++ni)
#pragma unroll
                for (int r = 0; r < 8; ++r) {
                    int grow = gm0 + wm * 32 + mi * 16 + r + hi * 8;
                    int gcol = gn0 + wn * 64 + ni * 16 + ln;
                    Cout[(size_t)grow * N + gcol] = acc[mi][ni][r];
                }
    } else {
        // N-chunk c: 0..15 -> q head c, 16..31 -> k head c-16, 32..47 -> v
        const int  c   = blockIdx.y;
        const bool isV = (c >= 32);
        if (tid < 128) {
            ssq[tid] = 0.f;
            gs[tid]  = isV ? 1.f : ((c < 16) ? qg[tid] : kg[tid]);
        }
        __syncthreads();
        if (!isV) {
#pragma unroll
            for (int mi = 0; mi < 2; ++mi)
#pragma unroll
                for (int ni = 0; ni < 4; ++ni)
#pragma unroll
                    for (int r = 0; r < 8; ++r) {
                        float v = acc[mi][ni][r];
                        atomicAdd(&ssq[wm * 32 + mi * 16 + r + hi * 8], v * v);
                    }
            __syncthreads();
        }
        const int h = (c < 16) ? c : ((c < 32) ? (c - 16) : (c - 32));
        __bf16* dstBase = (c < 16) ? qOut : ((c < 32) ? kOut : vOut);
#pragma unroll
        for (int mi = 0; mi < 2; ++mi)
#pragma unroll
            for (int r = 0; r < 8; ++r) {
                int rl = wm * 32 + mi * 16 + r + hi * 8;
                int gm = gm0 + rl;
                int bb = gm >> 11;        // token -> batch
                int s  = gm & 2047;       // token -> seq pos
                float scl = isV ? 1.f
                                : rsqrtf(ssq[rl] * (1.0f / 128.0f) + 1.1920929e-07f);
                __bf16* dp = dstBase + ((size_t)(bb * 16 + h) * 2048 + s) * 128;
#pragma unroll
                for (int ni = 0; ni < 4; ++ni) {
                    int col = wn * 64 + ni * 16 + ln;   // hd index
                    dp[col] = (__bf16)(acc[mi][ni][r] * scl * gs[col]);
                }
            }
    }
}

// ---------------------------------------------------------------------------
// Flash-style causal attention, bf16 WMMA, fp32 online softmax.
// Grid: (S/64, H, B). Block: 128 threads = 4 waves; wave owns 16 q-rows.
// K tile staged by the Tensor Data Mover (TENSOR_LOAD_TO_LDS, TENSORcnt);
// V tile transposed manually on store so PV B-fragments read contiguously.
// Per j-step (32 kv rows): scores = 8 wmma, PV = 8 wmma.
// ---------------------------------------------------------------------------
__global__ void __launch_bounds__(128)
attn_fwd(const __bf16* __restrict__ Q, const __bf16* __restrict__ Kmat,
         const __bf16* __restrict__ V, __bf16* __restrict__ Out)
{
    __shared__ __bf16 Ks[32 * 128];     // K tile, row-major [kv][hd]  (TDM dest)
    __shared__ __bf16 Vt[128 * 32];     // V tile transposed [hd][kv]
    __shared__ __bf16 Ps[4 * 16 * 32];  // per-wave P tile [m][kv]

    const int tid  = threadIdx.x;
    const int lane = tid & 31;
    const int wid  = tid >> 5;          // 0..3
    const int ln   = lane & 15;
    const int hi   = lane >> 4;
    const int b    = blockIdx.z;
    const int h    = blockIdx.y;
    const int qb   = blockIdx.x * 64;
    const int qw   = qb + wid * 16;
    const size_t hb = (size_t)(b * 16 + h) * 2048 * 128;

    const unsigned ksLds = (unsigned)(size_t)(&Ks[0]);   // LDS byte offset

    // Q fragments (A-layout), 16 rows x 128 hd, loaded once from global
    Frag qf[4];
    {
        const __bf16* qrow = Q + hb + (size_t)(qw + ln) * 128;
#pragma unroll
        for (int kk = 0; kk < 4; ++kk) {
            qf[kk].q[0] = *(const uint4*)(qrow + kk * 32 + hi * 8);
            qf[kk].q[1] = *(const uint4*)(qrow + kk * 32 + hi * 8 + 16);
        }
    }

    const f32x8 zero = {0.f, 0.f, 0.f, 0.f, 0.f, 0.f, 0.f, 0.f};
    f32x8 o[8];
#pragma unroll
    for (int t = 0; t < 8; ++t) o[t] = zero;
    float Mr[8], Lr[8];
#pragma unroll
    for (int r = 0; r < 8; ++r) { Mr[r] = -3.0e38f; Lr[r] = 0.f; }

    const float scale = 0.08838834764831845f;  // 1/sqrt(128)
    const int jmax = qb / 32 + 2;              // causal tile bound

    const int vKv = tid & 31, vSeg = tid >> 5; // V staging (transposed)

    for (int j = 0; j < jmax; ++j) {
        const int kv0 = j * 32;
        __syncthreads();   // prior-iteration readers of Ks/Vt are done
        if (wid == 0) {
            // DMA K tile [kv0 .. kv0+32) x 128 into LDS via the TDM
            tdm_load_tile_b16_128w(Kmat + hb + (size_t)kv0 * 128, ksLds, 32);
        }
        {
            union { uint4 q[4]; __bf16 e[32]; } vv;
            const uint4* vsrc = (const uint4*)(V + hb + (size_t)(kv0 + vKv) * 128 + vSeg * 32);
            vv.q[0] = vsrc[0]; vv.q[1] = vsrc[1]; vv.q[2] = vsrc[2]; vv.q[3] = vsrc[3];
#pragma unroll
            for (int e = 0; e < 32; ++e)
                Vt[(vSeg * 32 + e) * 32 + vKv] = vv.e[e];
        }
        if (wid == 0) __builtin_amdgcn_s_wait_tensorcnt(0);
        __syncthreads();   // publishes TDM K tile + Vt to all waves

        // S = Q K^T : two 16x16 tiles over this 32-kv chunk
        f32x8 sc[2] = {zero, zero};
#pragma unroll
        for (int nt = 0; nt < 2; ++nt)
#pragma unroll
            for (int kk = 0; kk < 4; ++kk) {
                Frag kb;
                const __bf16* p = Ks + (nt * 16 + ln) * 128 + kk * 32 + hi * 8;
                kb.q[0] = *(const uint4*)p;
                kb.q[1] = *(const uint4*)(p + 16);
                sc[nt] = wmma_bf16(qf[kk], kb, sc[nt]);
            }

        // causal mask + online softmax (rows live across 16-lane half-waves)
#pragma unroll
        for (int r = 0; r < 8; ++r) {
            const int qrow = qw + r + hi * 8;
            float s0 = sc[0][r] * scale;
            float s1 = sc[1][r] * scale;
            if (kv0 + ln      > qrow) s0 = -3.0e38f;
            if (kv0 + 16 + ln > qrow) s1 = -3.0e38f;
            float mt = fmaxf(s0, s1);
#pragma unroll
            for (int d = 1; d < 16; d <<= 1) mt = fmaxf(mt, __shfl_xor(mt, d, 32));
            const float mnew = fmaxf(Mr[r], mt);
            const float p0 = __expf(s0 - mnew);
            const float p1 = __expf(s1 - mnew);
            float lt = p0 + p1;
#pragma unroll
            for (int d = 1; d < 16; d <<= 1) lt += __shfl_xor(lt, d, 32);
            const float alpha = __expf(Mr[r] - mnew);
            Lr[r] = Lr[r] * alpha + lt;
            Mr[r] = mnew;
#pragma unroll
            for (int t = 0; t < 8; ++t) o[t][r] *= alpha;
            // stage P (bf16) for C-layout -> A-layout re-tiling
            __bf16* prow = Ps + (wid * 16 + r + hi * 8) * 32;
            prow[ln]      = (__bf16)p0;
            prow[16 + ln] = (__bf16)p1;
        }

        // O += P(16x32) @ V(32x128); LDS ops are in-order within a wave
        {
            Frag pa;
            const __bf16* prow = Ps + (wid * 16 + ln) * 32;
            pa.q[0] = *(const uint4*)(prow + hi * 8);
            pa.q[1] = *(const uint4*)(prow + hi * 8 + 16);
#pragma unroll
            for (int t = 0; t < 8; ++t) {
                Frag vb;
                const __bf16* p = Vt + (t * 16 + ln) * 32 + hi * 8;
                vb.q[0] = *(const uint4*)p;
                vb.q[1] = *(const uint4*)(p + 16);
                o[t] = wmma_bf16(pa, vb, o[t]);
            }
        }
    }

    // finalize: O /= L, write bf16 token-major [B*S, D] for the out-proj GEMM
#pragma unroll
    for (int r = 0; r < 8; ++r) {
        const float inv = 1.0f / Lr[r];
        const int srow = qw + r + hi * 8;
        __bf16* dp = Out + ((size_t)(b * 2048 + srow)) * 2048 + h * 128;
#pragma unroll
        for (int t = 0; t < 8; ++t)
            dp[t * 16 + ln] = (__bf16)(o[t][r] * inv);
    }
}

// ---------------------------------------------------------------------------
// Host-side pipeline
// ---------------------------------------------------------------------------
extern "C" void kernel_launch(void* const* d_in, const int* in_sizes, int n_in,
                              void* d_out, int out_size, void* d_ws, size_t ws_size,
                              hipStream_t stream) {
    (void)in_sizes; (void)n_in; (void)out_size; (void)ws_size;

    const float* x     = (const float*)d_in[0];  // [4,2048,2048]
    const float* w_in  = (const float*)d_in[1];  // [6144,2048]
    const float* w_out = (const float*)d_in[2];  // [2048,2048]
    const float* qg    = (const float*)d_in[3];  // [128]
    const float* kg    = (const float*)d_in[4];  // [128]
    float* out = (float*)d_out;

    char* ws = (char*)d_ws;
    __bf16* xb  = (__bf16*)(ws);              // 32 MB  x in bf16
    __bf16* wib = (__bf16*)(ws +  33554432);  // 24 MB  w_in bf16
    __bf16* wob = (__bf16*)(ws +  58720256);  //  8 MB  w_out bf16
    __bf16* qbf = (__bf16*)(ws +  67108864);  // 32 MB  q [B,H,S,HD]
    __bf16* kbf = (__bf16*)(ws + 100663296);  // 32 MB  k [B,H,S,HD]
    __bf16* vbf = (__bf16*)(ws + 134217728);  // 32 MB  v [B,H,S,HD]
    __bf16* ab  = xb;  // attention output aliases x (dead after GEMM1)

    conv_f32_bf16<<<2048, 256, 0, stream>>>(x,     xb,  8192 * 2048);
    conv_f32_bf16<<<2048, 256, 0, stream>>>(w_in,  wib, 6144 * 2048);
    conv_f32_bf16<<<1024, 256, 0, stream>>>(w_out, wob, 2048 * 2048);

    // QKV projection fused with QK-RMSNorm + head layout
    gemm_bf16_wmma<true><<<dim3(64, 48), 256, 0, stream>>>(
        xb, wib, nullptr, qbf, kbf, vbf, qg, kg, 8192, 6144, 2048);

    // causal flash attention (TDM-staged K tiles)
    attn_fwd<<<dim3(32, 16, 4), 128, 0, stream>>>(qbf, kbf, vbf, ab);

    // output projection, fp32 result
    gemm_bf16_wmma<false><<<dim3(64, 16), 256, 0, stream>>>(
        ab, wob, out, nullptr, nullptr, nullptr, nullptr, nullptr, 8192, 2048, 2048);
}
